// Mamba2Layer_27556510171315
// MI455X (gfx1250) — compile-verified, matrix operand order checked by reading, not run on hardware
//
#include <hip/hip_runtime.h>
#include <hip/hip_bf16.h>

// ---------------- model constants ----------------
#define D_MODEL   2048
#define D_INNER   4096
#define NHEADS    32
#define HEADDIM   128
#define DSTATE    64
#define CHUNK     256
#define NCHUNK    8
#define SEQLEN    2048
#define BATCH     4
#define CONV_DIM  4224          // D_INNER + 2*DSTATE
#define NPROJ     8352          // D_INNER + CONV_DIM + NHEADS
#define NPROJ_PAD 8448          // padded to multiple of 128 for the GEMM
#define M_TOK     8192          // BATCH*SEQLEN
#define DT_COL    8320          // D_INNER + CONV_DIM

typedef __attribute__((ext_vector_type(16))) __bf16 v16bf;
typedef __attribute__((ext_vector_type(8)))  __bf16 v8bf;
typedef __attribute__((ext_vector_type(8)))  float  v8f;

// Optional CDNA5 async global->LDS copy path (ASYNCcnt-tracked); falls back
// to synchronous VGPR staging if the builtins are not exposed by this clang.
#if defined(__has_builtin)
#if __has_builtin(__builtin_amdgcn_global_load_async_to_lds_b128) && \
    __has_builtin(__builtin_amdgcn_s_wait_asynccnt)
#define HAVE_ASYNC 1
#endif
#endif
#ifndef HAVE_ASYNC
#define HAVE_ASYNC 0
#endif

#if HAVE_ASYNC
// Parameter type per hipcc diagnostic: pointer to 16-byte int vector.
typedef int v4i_gl __attribute__((vector_size(4 * sizeof(int))));
#define ASYNC_CP16(gp, lp)                                                  \
  __builtin_amdgcn_global_load_async_to_lds_b128(                           \
      (v4i_gl*)(void*)const_cast<__bf16*>(gp), (v4i_gl*)(void*)(lp), 0, 0)
#endif

// Load a 16-bit A/B WMMA fragment from K-contiguous (row-major-in-K) memory.
// Per the CDNA5 16-bit operand layout, each lane needs two contiguous 8-elem
// runs: [half*8 .. half*8+7] and [16+half*8 .. 16+half*8+7].  Caller passes
// the pointer to the first run; second run is +16 elements.
__device__ __forceinline__ v16bf ldfrag(const __bf16* p) {
  v8bf a = *(const v8bf*)(p);
  v8bf b = *(const v8bf*)(p + 16);
  v16bf r;
#pragma unroll
  for (int i = 0; i < 8; ++i) { r[i] = a[i]; r[i + 8] = b[i]; }
  return r;
}

__device__ __forceinline__ v8f wmma_bf16(v16bf a, v16bf b, v8f c) {
  return __builtin_amdgcn_wmma_f32_16x16x32_bf16(false, a, false, b,
                                                 (short)0, c, false, false);
}

__device__ __forceinline__ float softplusf(float x) {
  return (x > 20.f) ? x : log1pf(__expf(x));
}
__device__ __forceinline__ float siluf(float x) {
  return x / (1.f + __expf(-x));
}

// ---------------- conversion kernels ----------------
__global__ void k_f32_to_bf16(const float* __restrict__ s,
                              __bf16* __restrict__ d, long n) {
  long i = (long)blockIdx.x * blockDim.x + threadIdx.x;
  if (i < n) d[i] = (__bf16)s[i];
}

// Convert W (rows_src x K) to bf16, zero-padding rows up to rows_dst.
__global__ void k_cvt_pad_rows(const float* __restrict__ s,
                               __bf16* __restrict__ d,
                               long rows_src, long n_total, int K) {
  long i = (long)blockIdx.x * blockDim.x + threadIdx.x;
  if (i >= n_total) return;
  long r = i / K;
  d[i] = (r < rows_src) ? (__bf16)s[i] : (__bf16)0.f;
}

// ---------------- bf16 WMMA GEMM:  C[M,N] = A[M,K] * B[N,K]^T ----------------
// Block tile 128x128, 8 waves (wave32), wave tile 64x32, K-step 32.
// Double-buffered LDS staging; next K-tile is fetched (async when available)
// while the current tile feeds the WMMAs.
__global__ __launch_bounds__(256) void k_gemm_bf16(
    const __bf16* __restrict__ A,   // M x K, row-major
    const __bf16* __restrict__ B,   // N x K, row-major (i.e. B^T of math B)
    float* __restrict__ C, int M, int N, int K, int ldc) {
  __shared__ __align__(16) __bf16 As[2][128 * 40];
  __shared__ __align__(16) __bf16 Bs[2][128 * 40];
  const int bm = blockIdx.y * 128, bn = blockIdx.x * 128;
  const int tid = threadIdx.x;
  const int w = tid >> 5, lane = tid & 31, half = lane >> 4, l16 = lane & 15;
  const int wm = (w >> 2) * 64;   // 2 wave rows
  const int wn = (w & 3) * 32;    // 4 wave cols
  // Each thread stages two 16B segments of each 128x32 tile (512 segs total).
  const int seg0 = tid * 2, r0 = seg0 >> 2, c0 = (seg0 & 3) * 8;
  const int seg1 = tid * 2 + 1, r1 = seg1 >> 2, c1 = (seg1 & 3) * 8;

  auto stage = [&](int buf, int k0) {
#if HAVE_ASYNC
    ASYNC_CP16(&A[(long)(bm + r0) * K + k0 + c0], &As[buf][r0 * 40 + c0]);
    ASYNC_CP16(&A[(long)(bm + r1) * K + k0 + c1], &As[buf][r1 * 40 + c1]);
    ASYNC_CP16(&B[(long)(bn + r0) * K + k0 + c0], &Bs[buf][r0 * 40 + c0]);
    ASYNC_CP16(&B[(long)(bn + r1) * K + k0 + c1], &Bs[buf][r1 * 40 + c1]);
#else
    *(v8bf*)&As[buf][r0 * 40 + c0] = *(const v8bf*)&A[(long)(bm + r0) * K + k0 + c0];
    *(v8bf*)&As[buf][r1 * 40 + c1] = *(const v8bf*)&A[(long)(bm + r1) * K + k0 + c1];
    *(v8bf*)&Bs[buf][r0 * 40 + c0] = *(const v8bf*)&B[(long)(bn + r0) * K + k0 + c0];
    *(v8bf*)&Bs[buf][r1 * 40 + c1] = *(const v8bf*)&B[(long)(bn + r1) * K + k0 + c1];
#endif
  };

  v8f acc[4][2] = {};
  const int nk = K / 32;
  stage(0, 0);
#pragma unroll 1
  for (int t = 0; t < nk; ++t) {
#if HAVE_ASYNC
    __builtin_amdgcn_s_wait_asynccnt(0);  // my tile-t segments are in LDS
#endif
    __syncthreads();                      // everyone's tile-t segments visible
    if (t + 1 < nk) stage((t + 1) & 1, (t + 1) * 32);
    const __bf16* Ab = &As[t & 1][0];
    const __bf16* Bb = &Bs[t & 1][0];
    v16bf af[4], bfr[2];
#pragma unroll
    for (int i = 0; i < 4; ++i)
      af[i] = ldfrag(&Ab[(wm + i * 16 + l16) * 40 + half * 8]);
#pragma unroll
    for (int j = 0; j < 2; ++j)
      bfr[j] = ldfrag(&Bb[(wn + j * 16 + l16) * 40 + half * 8]);
#pragma unroll
    for (int i = 0; i < 4; ++i)
#pragma unroll
      for (int j = 0; j < 2; ++j)
        acc[i][j] = wmma_bf16(af[i], bfr[j], acc[i][j]);
  }
  // D layout: lane n = l16, row m = vgpr + half*8
#pragma unroll
  for (int i = 0; i < 4; ++i)
#pragma unroll
    for (int j = 0; j < 2; ++j)
#pragma unroll
      for (int v = 0; v < 8; ++v) {
        long m = bm + wm + i * 16 + half * 8 + v;
        long n = bn + wn + j * 16 + l16;
        C[m * ldc + n] = acc[i][j][v];
      }
}

// ---------------- conv1d(width4) + SiLU + split + dt/xd ----------------
__global__ void k_conv_silu(const float* __restrict__ proj,   // [M_TOK, NPROJ_PAD]
                            const float* __restrict__ conv_w, // [CONV_DIM,4]
                            const float* __restrict__ conv_b,
                            const float* __restrict__ dt_bias,
                            float* __restrict__ xs,           // [M_TOK, D_INNER]
                            __bf16* __restrict__ xdT,         // [b][c][h][p][q]
                            __bf16* __restrict__ Bg,          // [M_TOK, 64]
                            __bf16* __restrict__ Cg) {        // [M_TOK, 64]
  long idx = (long)blockIdx.x * blockDim.x + threadIdx.x;
  if (idx >= (long)M_TOK * CONV_DIM) return;
  int ch = (int)(idx % CONV_DIM);
  long t = idx / CONV_DIM;
  int b = (int)(t / SEQLEN), l = (int)(t % SEQLEN);
  float acc = conv_b[ch];
#pragma unroll
  for (int k = 0; k < 4; ++k) {
    int ls = l + k - 3;
    float xv = (ls >= 0)
                   ? proj[((long)b * SEQLEN + ls) * NPROJ_PAD + D_INNER + ch]
                   : 0.f;
    acc += xv * conv_w[ch * 4 + k];
  }
  float v = siluf(acc);
  if (ch < D_INNER) {
    int h = ch >> 7, p = ch & 127;
    float dtr = proj[t * NPROJ_PAD + DT_COL + h] + dt_bias[h];
    float dtv = softplusf(dtr);
    xs[t * D_INNER + ch] = v;
    int c = l >> 8, q = l & 255;
    xdT[((((long)b * NCHUNK + c) * NHEADS + h) * HEADDIM + p) * CHUNK + q] =
        (__bf16)(v * dtv);
  } else if (ch < D_INNER + DSTATE) {
    Bg[t * DSTATE + (ch - D_INNER)] = (__bf16)v;
  } else {
    Cg[t * DSTATE + (ch - D_INNER - DSTATE)] = (__bf16)v;
  }
}

// ---------------- per-chunk cumulative sum of dt*A ----------------
__global__ void k_chunk_cumsum(const float* __restrict__ proj,
                               const float* __restrict__ dt_bias,
                               const float* __restrict__ A_log,
                               float* __restrict__ Acs) { // [b*c][h][q]
  int bc = blockIdx.x;                 // b*NCHUNK + c
  int b = bc >> 3, c = bc & 7;
  int h = threadIdx.x;
  if (h >= NHEADS) return;
  float Ah = -__expf(A_log[h]);
  float bias = dt_bias[h];
  float acc = 0.f;
  long row0 = (long)b * SEQLEN + c * CHUNK;
  long out0 = ((long)bc * NHEADS + h) * CHUNK;
  for (int q = 0; q < CHUNK; ++q) {
    float dtv = softplusf(proj[(row0 + q) * NPROJ_PAD + DT_COL + h] + bias);
    acc += dtv * Ah;
    Acs[out0 + q] = acc;
  }
}

// ---------------- SSD intra-chunk: Y_diag + chunk states ----------------
// One block per (b,c).  Computes CB = C*B^T (256x256) once with WMMA into
// LDS (bf16), then loops 32 heads: Y_diag = (CB .* L) @ xd  and
// states = (B .* decay)^T @ xd, both via WMMA.
__global__ __launch_bounds__(256) void k_ssd_intra(
    const __bf16* __restrict__ Bg,  // [M_TOK,64]
    const __bf16* __restrict__ Cg,  // [M_TOK,64]
    const __bf16* __restrict__ xdT, // [b][c][h][p][q]
    const float* __restrict__ Acs,  // [b*c][h][q]
    float* __restrict__ yb,         // [M_TOK, D_INNER]  (Y_diag out)
    float* __restrict__ states) {   // [b*c][h][n][p]
  __shared__ __align__(16) __bf16 Cs[256 * 72];
  __shared__ __align__(16) __bf16 Bls[256 * 72];
  __shared__ __align__(16) __bf16 CBs[256 * 264];
  __shared__ float acs_sh[256];

  const int bc = blockIdx.x;
  const int b = bc >> 3, c = bc & 7;
  const int tid = threadIdx.x;
  const int w = tid >> 5, lane = tid & 31, half = lane >> 4, l16 = lane & 15;
  const long tok0 = (long)b * SEQLEN + c * CHUNK;

  // Stage C (256x64) and B (256x64), K(=n)-contiguous.
  {
    const v8bf* sC = (const v8bf*)&Cg[(tok0 + tid) * DSTATE];
    const v8bf* sB = (const v8bf*)&Bg[(tok0 + tid) * DSTATE];
    v8bf* dC = (v8bf*)&Cs[tid * 72];
    v8bf* dB = (v8bf*)&Bls[tid * 72];
#pragma unroll
    for (int i = 0; i < 8; ++i) { dC[i] = sC[i]; dB[i] = sB[i]; }
  }
  __syncthreads();

  // ---- CB[q][k] = sum_n C[q][n]*B[k][n]  (M=256,N=256,K=64) ----
  {
    int mb = w * 32; // each wave owns 32 M rows
#pragma unroll 1
    for (int nb = 0; nb < 256; nb += 64) {
      v8f acc[2][4] = {};
#pragma unroll
      for (int ks = 0; ks < 64; ks += 32) {
        v16bf af[2], bfr[4];
#pragma unroll
        for (int i = 0; i < 2; ++i)
          af[i] = ldfrag(&Cs[(mb + i * 16 + l16) * 72 + ks + half * 8]);
#pragma unroll
        for (int j = 0; j < 4; ++j)
          bfr[j] = ldfrag(&Bls[(nb + j * 16 + l16) * 72 + ks + half * 8]);
#pragma unroll
        for (int i = 0; i < 2; ++i)
#pragma unroll
          for (int j = 0; j < 4; ++j)
            acc[i][j] = wmma_bf16(af[i], bfr[j], acc[i][j]);
      }
#pragma unroll
      for (int i = 0; i < 2; ++i)
#pragma unroll
        for (int j = 0; j < 4; ++j)
#pragma unroll
          for (int v = 0; v < 8; ++v)
            CBs[(mb + i * 16 + half * 8 + v) * 264 + nb + j * 16 + l16] =
                (__bf16)acc[i][j][v];
    }
  }

  // ---- per-head Y_diag + states ----
#pragma unroll 1
  for (int h = 0; h < NHEADS; ++h) {
    __syncthreads();
    acs_sh[tid] = Acs[((long)bc * NHEADS + h) * CHUNK + tid];
    __syncthreads();
    const float acs_last = acs_sh[255];
    const __bf16* xdh =
        xdT + (((long)bc * NHEADS + h) * HEADDIM) * CHUNK; // [p][q], ld=256

    // Y_diag: M=256(q), N=128(p), K=256(k); wave tile 64x64 (grid 4x2)
    {
      int wm = (w >> 1) * 64, wn = (w & 1) * 64;
      v8f acc[4][4] = {};
#pragma unroll 1
      for (int ks = 0; ks < 256; ks += 32) {
        v16bf bfr[4];
#pragma unroll
        for (int j = 0; j < 4; ++j)
          bfr[j] = ldfrag(&xdh[(wn + j * 16 + l16) * CHUNK + ks + half * 8]);
#pragma unroll
        for (int i = 0; i < 4; ++i) {
          int q = wm + i * 16 + l16;
          float eq = acs_sh[q];
          v8bf c0 = *(const v8bf*)&CBs[q * 264 + ks + half * 8];
          v8bf c1 = *(const v8bf*)&CBs[q * 264 + ks + 16 + half * 8];
          v16bf af;
#pragma unroll
          for (int e = 0; e < 8; ++e) {
            int k1 = ks + half * 8 + e;
            int k2 = ks + 16 + half * 8 + e;
            float v1 = (k1 <= q) ? (float)c0[e] * __expf(eq - acs_sh[k1]) : 0.f;
            float v2 = (k2 <= q) ? (float)c1[e] * __expf(eq - acs_sh[k2]) : 0.f;
            af[e] = (__bf16)v1;
            af[e + 8] = (__bf16)v2;
          }
#pragma unroll
          for (int j = 0; j < 4; ++j)
            acc[i][j] = wmma_bf16(af, bfr[j], acc[i][j]);
        }
      }
#pragma unroll
      for (int i = 0; i < 4; ++i)
#pragma unroll
        for (int j = 0; j < 4; ++j)
#pragma unroll
          for (int v = 0; v < 8; ++v) {
            int q = wm + i * 16 + half * 8 + v;
            int p = wn + j * 16 + l16;
            yb[(tok0 + q) * D_INNER + h * HEADDIM + p] = acc[i][j][v];
          }
    }

    // states: S[n][p] = sum_k B[k][n]*exp(acs_last-acs[k])*xd[k][p]
    // M=64(n), N=128(p), K=256; wave tile 64x16 (8 waves across N)
    {
      int wn2 = w * 16;
      v8f acc2[4] = {};
#pragma unroll 1
      for (int ks = 0; ks < 256; ks += 32) {
        v16bf bfr = ldfrag(&xdh[(wn2 + l16) * CHUNK + ks + half * 8]);
#pragma unroll
        for (int i = 0; i < 4; ++i) {
          int nrow = i * 16 + l16;
          v16bf af;
#pragma unroll
          for (int e = 0; e < 8; ++e) {
            int k1 = ks + half * 8 + e;
            int k2 = ks + 16 + half * 8 + e;
            af[e] = (__bf16)((float)Bls[k1 * 72 + nrow] *
                             __expf(acs_last - acs_sh[k1]));
            af[e + 8] = (__bf16)((float)Bls[k2 * 72 + nrow] *
                                 __expf(acs_last - acs_sh[k2]));
          }
          acc2[i] = wmma_bf16(af, bfr, acc2[i]);
        }
      }
#pragma unroll
      for (int i = 0; i < 4; ++i)
#pragma unroll
        for (int v = 0; v < 8; ++v) {
          int nrow = i * 16 + half * 8 + v;
          states[(((long)bc * NHEADS + h) * DSTATE + nrow) * HEADDIM + wn2 +
                 l16] = acc2[i][v];
        }
    }
  }
}

// ---------------- inter-chunk state scan (sequential over chunks) ----------
__global__ __launch_bounds__(256) void k_scan(
    const float* __restrict__ states, // [b*c][h][n][p]
    const float* __restrict__ Acs,    // [b*c][h][q]
    __bf16* __restrict__ prevT) {     // [b*c][h][p][n]  (transposed, bf16)
  int bh = blockIdx.x; // b*NHEADS + h
  int b = bh >> 5, h = bh & 31;
  int tid = threadIdx.x;
  float carry[32];
#pragma unroll
  for (int i = 0; i < 32; ++i) carry[i] = 0.f;
#pragma unroll 1
  for (int c = 0; c < NCHUNK; ++c) {
    long base = ((long)(b * NCHUNK + c) * NHEADS + h);
    float cd = __expf(Acs[base * CHUNK + 255]);
#pragma unroll
    for (int i = 0; i < 32; ++i) {
      int idx = tid + i * 256; // over 64*128
      int n = idx >> 7, p = idx & 127;
      prevT[base * (DSTATE * HEADDIM) + p * DSTATE + n] = (__bf16)carry[i];
      carry[i] = carry[i] * cd + states[base * (DSTATE * HEADDIM) + idx];
    }
  }
}

// ---------------- Y_off = exp(Acs[q]) * C @ prev, plus D-skip -------------
// One block per (b,c,h).  M=256(q), N=128(p), K=64(n).
__global__ __launch_bounds__(256) void k_ssd_off(
    const __bf16* __restrict__ Cg,    // [M_TOK,64]
    const __bf16* __restrict__ prevT, // [b*c][h][p][n]
    const float* __restrict__ Acs,    // [b*c][h][q]
    const float* __restrict__ xs,     // [M_TOK, D_INNER]
    const float* __restrict__ Dp,     // [NHEADS]
    float* __restrict__ yb) {         // in/out [M_TOK, D_INNER]
  int bch = blockIdx.x;
  int bc = bch >> 5, h = bch & 31;
  int b = bc >> 3, c = bc & 7;
  int tid = threadIdx.x, w = tid >> 5, lane = tid & 31;
  int half = lane >> 4, l16 = lane & 15;
  long tok0 = (long)b * SEQLEN + c * CHUNK;
  long base = (long)bc * NHEADS + h;
  float Dh = Dp[h];
  int wm = (w >> 1) * 64, wn = (w & 1) * 64;
  v8f acc[4][4] = {};
#pragma unroll
  for (int ks = 0; ks < 64; ks += 32) {
    v16bf af[4], bfr[4];
#pragma unroll
    for (int i = 0; i < 4; ++i)
      af[i] = ldfrag(&Cg[(tok0 + wm + i * 16 + l16) * DSTATE + ks + half * 8]);
#pragma unroll
    for (int j = 0; j < 4; ++j)
      bfr[j] = ldfrag(&prevT[base * (DSTATE * HEADDIM) +
                             (wn + j * 16 + l16) * DSTATE + ks + half * 8]);
#pragma unroll
    for (int i = 0; i < 4; ++i)
#pragma unroll
      for (int j = 0; j < 4; ++j)
        acc[i][j] = wmma_bf16(af[i], bfr[j], acc[i][j]);
  }
#pragma unroll
  for (int i = 0; i < 4; ++i)
#pragma unroll
    for (int j = 0; j < 4; ++j)
#pragma unroll
      for (int v = 0; v < 8; ++v) {
        int q = wm + i * 16 + half * 8 + v;
        int p = wn + j * 16 + l16;
        float e = __expf(Acs[base * CHUNK + q]);
        long o = (tok0 + q) * D_INNER + h * HEADDIM + p;
        yb[o] = yb[o] + acc[i][j][v] * e + Dh * xs[o];
      }
}

// ---------------- RMS-norm + SiLU(z) gate, emit bf16 for GEMM2 ------------
__global__ __launch_bounds__(256) void k_norm_gate(
    const float* __restrict__ yb,   // [M_TOK, D_INNER]
    const float* __restrict__ proj, // z = proj[:, 0:D_INNER]
    __bf16* __restrict__ ygb) {     // [M_TOK, D_INNER]
  __shared__ float red[256];
  long row = blockIdx.x;
  const float* yr = yb + row * D_INNER;
  float ss = 0.f;
#pragma unroll
  for (int i = 0; i < 16; ++i) {
    float v = yr[threadIdx.x + i * 256];
    ss += v * v;
  }
  red[threadIdx.x] = ss;
  __syncthreads();
  for (int s = 128; s > 0; s >>= 1) {
    if ((int)threadIdx.x < s) red[threadIdx.x] += red[threadIdx.x + s];
    __syncthreads();
  }
  float rms = rsqrtf(red[0] / (float)D_INNER + 1.1920929e-07f);
  const float* zr = proj + row * NPROJ_PAD;
#pragma unroll
  for (int i = 0; i < 16; ++i) {
    int idx = threadIdx.x + i * 256;
    float g = siluf(zr[idx]);
    ygb[row * D_INNER + idx] = (__bf16)(yr[idx] * rms * g);
  }
}

// ---------------- host orchestration ----------------
extern "C" void kernel_launch(void* const* d_in, const int* in_sizes, int n_in,
                              void* d_out, int out_size, void* d_ws,
                              size_t ws_size, hipStream_t stream) {
  const float* x       = (const float*)d_in[0];
  const float* W_in    = (const float*)d_in[1];
  const float* conv_w  = (const float*)d_in[2];
  const float* conv_b  = (const float*)d_in[3];
  const float* A_log   = (const float*)d_in[4];
  const float* dt_bias = (const float*)d_in[5];
  const float* Dp      = (const float*)d_in[6];
  const float* W_out   = (const float*)d_in[7];
  (void)in_sizes; (void)n_in; (void)out_size; (void)ws_size;

  char* ws = (char*)d_ws;
  size_t off = 0;
  auto alloc = [&](size_t bytes) {
    size_t p = off;
    off += (bytes + 255) & ~(size_t)255;
    return p;
  };
  __bf16* xb    = (__bf16*)(ws + alloc((size_t)M_TOK * D_MODEL * 2));
  __bf16* wb    = (__bf16*)(ws + alloc((size_t)NPROJ_PAD * D_MODEL * 2));
  float*  proj  = (float*) (ws + alloc((size_t)M_TOK * NPROJ_PAD * 4));
  float*  xs    = (float*) (ws + alloc((size_t)M_TOK * D_INNER * 4));
  __bf16* xdT   = (__bf16*)(ws + alloc((size_t)M_TOK * D_INNER * 2));
  __bf16* Bg    = (__bf16*)(ws + alloc((size_t)M_TOK * DSTATE * 2));
  __bf16* Cg    = (__bf16*)(ws + alloc((size_t)M_TOK * DSTATE * 2));
  float*  Acs   = (float*) (ws + alloc((size_t)BATCH * NCHUNK * NHEADS * CHUNK * 4));
  float*  yb    = (float*) (ws + alloc((size_t)M_TOK * D_INNER * 4));
  float*  st    = (float*) (ws + alloc((size_t)BATCH * NCHUNK * NHEADS * DSTATE * HEADDIM * 4));
  __bf16* prevT = (__bf16*)(ws + alloc((size_t)BATCH * NCHUNK * NHEADS * DSTATE * HEADDIM * 2));
  __bf16* ygb   = (__bf16*)(ws + alloc((size_t)M_TOK * D_INNER * 2));
  __bf16* wob   = (__bf16*)(ws + alloc((size_t)D_MODEL * D_INNER * 2));

  // 1) convert operands to bf16 (pad W_in rows 8352 -> 8448 with zeros)
  {
    long n = (long)M_TOK * D_MODEL;
    k_f32_to_bf16<<<(unsigned)((n + 255) / 256), 256, 0, stream>>>(x, xb, n);
  }
  {
    long n = (long)NPROJ_PAD * D_MODEL;
    k_cvt_pad_rows<<<(unsigned)((n + 255) / 256), 256, 0, stream>>>(
        W_in, wb, NPROJ, n, D_MODEL);
  }
  {
    long n = (long)D_MODEL * D_INNER;
    k_f32_to_bf16<<<(unsigned)((n + 255) / 256), 256, 0, stream>>>(W_out, wob, n);
  }

  // 2) in-projection GEMM: proj[8192,8448] = xb[8192,2048] * wb[8448,2048]^T
  k_gemm_bf16<<<dim3(NPROJ_PAD / 128, M_TOK / 128), 256, 0, stream>>>(
      xb, wb, proj, M_TOK, NPROJ_PAD, D_MODEL, NPROJ_PAD);

  // 3) causal conv1d + SiLU + split (+ dt softplus, xd^T bf16)
  {
    long n = (long)M_TOK * CONV_DIM;
    k_conv_silu<<<(unsigned)((n + 255) / 256), 256, 0, stream>>>(
        proj, conv_w, conv_b, dt_bias, xs, xdT, Bg, Cg);
  }

  // 4) per-chunk cumsum of dt*A
  k_chunk_cumsum<<<BATCH * NCHUNK, 32, 0, stream>>>(proj, dt_bias, A_log, Acs);

  // 5) intra-chunk SSD (Y_diag + states), WMMA-heavy
  k_ssd_intra<<<BATCH * NCHUNK, 256, 0, stream>>>(Bg, Cg, xdT, Acs, yb, st);

  // 6) inter-chunk sequential scan
  k_scan<<<BATCH * NHEADS, 256, 0, stream>>>(st, Acs, prevT);

  // 7) Y_off + D-skip
  k_ssd_off<<<BATCH * NCHUNK * NHEADS, 256, 0, stream>>>(Cg, prevT, Acs, xs,
                                                         Dp, yb);

  // 8) RMS-norm + SiLU(z) gate -> bf16
  k_norm_gate<<<M_TOK, 256, 0, stream>>>(yb, proj, ygb);

  // 9) out-projection GEMM: out[8192,2048] = ygb[8192,4096] * wob[2048,4096]^T
  k_gemm_bf16<<<dim3(D_MODEL / 128, M_TOK / 128), 256, 0, stream>>>(
      ygb, wob, (float*)d_out, M_TOK, D_MODEL, D_INNER, D_MODEL);
}
